// RNN_10299331575913
// MI455X (gfx1250) — compile-verified, matrix operand order checked by reading, not run on hardware
//
#include <hip/hip_runtime.h>
#include <math.h>

// ---------------- problem constants ----------------
#define BB     128
#define SEQT   500
#define INP    306
#define INPAD  320      // 306 padded to multiple of 32 (one WMMA K-tile)
#define HD     512
#define NCLS   4
#define BTROWS (BB * SEQT)   // 64000 GEMM rows

typedef __attribute__((ext_vector_type(16))) __bf16 v16bf;
typedef __attribute__((ext_vector_type(8)))  float  v8f;
typedef __attribute__((ext_vector_type(4)))  unsigned int u32x4;
typedef __attribute__((ext_vector_type(8)))  int    i32x8;
typedef __attribute__((ext_vector_type(4)))  int    i32x4;

// ---------------- WMMA helpers ----------------
__device__ __forceinline__ v8f wmma_bf16(v16bf a, v16bf b, v8f c) {
  return __builtin_amdgcn_wmma_f32_16x16x32_bf16(
      /*neg_a=*/false, a, /*neg_b=*/false, b,
      /*c_mod=*/(short)0, c, /*reuse_a=*/false, /*reuse_b=*/false);
}

// Load one 16x32 bf16 fragment per the CDNA5 16-bit A/B VGPR layout:
// lane = row (M or N), lane[4] selects K-half; two 16B loads per lane.
__device__ __forceinline__ v16bf load_frag(const __bf16* __restrict__ base, int ld) {
  const int lane = threadIdx.x & 31;
  const int row  = lane & 15;
  const int kh   = (lane >> 4) << 3;       // 0 or 8
  const __bf16* p = base + (size_t)row * (size_t)ld + kh;
  union { v16bf v; uint4 u[2]; } t;
  t.u[0] = *reinterpret_cast<const uint4*>(p);
  t.u[1] = *reinterpret_cast<const uint4*>(p + 16);
  return t.v;
}

// ---------------- Tensor Data Mover: 2D fp32 tile -> LDS --------------------
// Loads a (rows=16) x (cols=512) fp32 tile whose rows are strided by
// row_stride_elems in global memory, packed contiguously into LDS at
// lds_byte_off. D# per CDNA5 ISA.
__device__ __forceinline__ void tdm_load_pre_tile(const float* gptr,
                                                  unsigned lds_byte_off,
                                                  unsigned row_stride_elems) {
  unsigned long long ga = (unsigned long long)(size_t)gptr;
  u32x4 g0;
  g0[0] = 1u;                                          // count=1 (valid user D#)
  g0[1] = lds_byte_off;                                // lds_addr [63:32]
  g0[2] = (unsigned)(ga & 0xFFFFFFFFu);                // global_addr lo
  g0[3] = (unsigned)((ga >> 32) & 0x01FFFFFFu)         // global_addr [56:32]
        | 0x80000000u;                                 // type=2 ("image") [127:126]
  i32x8 g1;
  g1[0] = (int)(2u << 16);          // wg_mask=0 | data_size=2 (4 bytes)
  g1[1] = (int)(512u << 16);        // tensor_dim0[15:0]=512  (bits 63:48)
  g1[2] = (int)(16u  << 16);        // tensor_dim0 hi=0 | tensor_dim1[15:0]=16
  g1[3] = (int)(512u << 16);        // tensor_dim1 hi=0 | tile_dim0=512
  g1[4] = (int)16;                  // tile_dim1=16 | tile_dim2=0
  g1[5] = (int)row_stride_elems;    // tensor_dim0_stride lo (48-bit)
  g1[6] = 0;                        // stride hi | tensor_dim1_stride lo
  g1[7] = 0;                        // tensor_dim1_stride hi
  i32x4 z4 = {0, 0, 0, 0};
#if __clang_major__ >= 23
  i32x8 z8 = {0, 0, 0, 0, 0, 0, 0, 0};
  __builtin_amdgcn_tensor_load_to_lds(g0, g1, z4, z4, z8, 0);
#else
  __builtin_amdgcn_tensor_load_to_lds(g0, g1, z4, z4, 0);
#endif
}

// ---------------- pack kernels (fp32 -> bf16 with K padding) ----------------
__global__ void pack_bf16_pad(const float* __restrict__ src, __bf16* __restrict__ dst,
                              int rows, int cols_src, int cols_dst) {
  int idx = blockIdx.x * blockDim.x + threadIdx.x;
  if (idx >= rows * cols_dst) return;
  int r = idx / cols_dst;
  int c = idx - r * cols_dst;
  float v = (c < cols_src) ? src[(size_t)r * cols_src + c] : 0.0f;
  dst[idx] = (__bf16)v;
}

// ---------------- batched GEMM: out[M][512] = A[M][K]*(W[N][K])^T + b0+b1 ----
// block = 256 threads = 8 waves arranged 4(M) x 2(N); wave tile = 32x32
__global__ void gemm_bias(const __bf16* __restrict__ A,
                          const __bf16* __restrict__ W,
                          const float* __restrict__ bias0,
                          const float* __restrict__ bias1,
                          float* __restrict__ out,
                          int ldk) {                 // lda == ldb == K
  const int wave = threadIdx.x >> 5;
  const int lane = threadIdx.x & 31;
  const int mw = wave & 3, nw = wave >> 2;
  const int mbase = blockIdx.x * 128 + mw * 32;
  const int nbase = blockIdx.y * 64  + nw * 32;

  v8f acc[2][2] = {};
  for (int kb = 0; kb < ldk; kb += 32) {
    v16bf a0 = load_frag(A + (size_t)(mbase +  0) * ldk + kb, ldk);
    v16bf a1 = load_frag(A + (size_t)(mbase + 16) * ldk + kb, ldk);
    v16bf b0 = load_frag(W + (size_t)(nbase +  0) * ldk + kb, ldk);
    v16bf b1 = load_frag(W + (size_t)(nbase + 16) * ldk + kb, ldk);
    acc[0][0] = wmma_bf16(a0, b0, acc[0][0]);
    acc[0][1] = wmma_bf16(a0, b1, acc[0][1]);
    acc[1][0] = wmma_bf16(a1, b0, acc[1][0]);
    acc[1][1] = wmma_bf16(a1, b1, acc[1][1]);
  }
#pragma unroll
  for (int i = 0; i < 2; ++i) {
#pragma unroll
    for (int j = 0; j < 2; ++j) {
      int n = nbase + j * 16 + (lane & 15);
      float bsum = bias0[n] + bias1[n];
#pragma unroll
      for (int r = 0; r < 8; ++r) {
        int m = mbase + i * 16 + ((lane >> 4) << 3) + r;
        out[(size_t)m * HD + n] = acc[i][j][r] + bsum;
      }
    }
  }
}

// ---------------- recurrent scan: h_t = tanh(pre_t + h_{t-1} * Whh^T) -------
// One WG per 16 batch rows (independent recurrences -> no inter-WG sync).
// h in LDS ping-pong (32KB); pre_t tile TDM-prefetched one step ahead into an
// LDS ping-pong (64KB) so the epilogue reads LDS, not strided global fp32.
__global__ void rnn_scan(const float*  __restrict__ pre,   // [B][T][H], row = b*T+t
                         const __bf16* __restrict__ W,     // [H][H] bf16
                         __bf16*       __restrict__ out,   // store_all ? [B][T][H] : [B][H]
                         int store_all) {
  __shared__ __align__(16) __bf16 hbuf[2][16 * HD];   // 32 KB
  __shared__ __align__(64) float  ptile[2][16 * HD];  // 64 KB
  const int b0   = blockIdx.x * 16;
  const int wave = threadIdx.x >> 5;
  const int lane = threadIdx.x & 31;
  const int nb   = wave * 64;                 // this wave's 64 output columns

  for (int i = threadIdx.x; i < 16 * HD; i += blockDim.x)
    hbuf[0][i] = (__bf16)0.0f;

  // prime the pipeline: DMA pre tile for t=0
  if (wave == 0) {
    tdm_load_pre_tile(pre + (size_t)b0 * SEQT * HD,
                      (unsigned)(size_t)&ptile[0][0], SEQT * HD);
    __builtin_amdgcn_s_wait_tensorcnt(0);
  }
  __syncthreads();

  int cur = 0;
  for (int t = 0; t < SEQT; ++t) {
    // overlap next step's pre DMA with this step's WMMA chain
    if (wave == 0 && (t + 1) < SEQT)
      tdm_load_pre_tile(pre + ((size_t)b0 * SEQT + (t + 1)) * HD,
                        (unsigned)(size_t)&ptile[(t + 1) & 1][0], SEQT * HD);

    v8f acc[4] = {};
    for (int kb = 0; kb < HD; kb += 32) {
      v16bf a = load_frag(&hbuf[cur][kb], HD);               // ds_read_b128 x2
#pragma unroll
      for (int j = 0; j < 4; ++j) {
        v16bf b = load_frag(W + (size_t)(nb + j * 16) * HD + kb, HD);
        acc[j] = wmma_bf16(a, b, acc[j]);
      }
    }
#pragma unroll
    for (int j = 0; j < 4; ++j) {
      int n = nb + j * 16 + (lane & 15);
#pragma unroll
      for (int r = 0; r < 8; ++r) {
        int m = ((lane >> 4) << 3) + r;
        size_t row = (size_t)(b0 + m) * SEQT + t;
        float v = acc[j][r] + ptile[t & 1][m * HD + n];      // ds_read_b32
        __bf16 hb = (__bf16)tanhf(v);
        hbuf[cur ^ 1][m * HD + n] = hb;
        if (store_all)            out[row * HD + n] = hb;
        else if (t == SEQT - 1)   out[(size_t)(b0 + m) * HD + n] = hb;
      }
    }
    // make sure next step's pre tile has landed before anyone proceeds
    if (wave == 0 && (t + 1) < SEQT)
      __builtin_amdgcn_s_wait_tensorcnt(0);
    __syncthreads();
    cur ^= 1;
  }
}

// ---------------- final FC + softmax (tiny) --------------------------------
__global__ void fc_softmax(const __bf16* __restrict__ hlast,  // [B][H]
                           const float* __restrict__ Wfc,     // [NCLS][H]
                           const float* __restrict__ bfc,
                           float* __restrict__ out) {         // [B][NCLS]
  int b = blockIdx.x * blockDim.x + threadIdx.x;
  if (b >= BB) return;
  float logit[NCLS];
#pragma unroll
  for (int c = 0; c < NCLS; ++c) {
    float s = bfc[c];
    for (int h = 0; h < HD; ++h)
      s += (float)hlast[(size_t)b * HD + h] * Wfc[(size_t)c * HD + h];
    logit[c] = s;
  }
  float mx = logit[0];
#pragma unroll
  for (int c = 1; c < NCLS; ++c) mx = fmaxf(mx, logit[c]);
  float sum = 0.0f, e[NCLS];
#pragma unroll
  for (int c = 0; c < NCLS; ++c) { e[c] = __expf(logit[c] - mx); sum += e[c]; }
  float inv = 1.0f / sum;
#pragma unroll
  for (int c = 0; c < NCLS; ++c) out[(size_t)b * NCLS + c] = e[c] * inv;
}

// ---------------- launcher -------------------------------------------------
extern "C" void kernel_launch(void* const* d_in, const int* in_sizes, int n_in,
                              void* d_out, int out_size, void* d_ws, size_t ws_size,
                              hipStream_t stream) {
  const float* x    = (const float*)d_in[0];
  const float* Wih0 = (const float*)d_in[1];
  const float* Whh0 = (const float*)d_in[2];
  const float* bih0 = (const float*)d_in[3];
  const float* bhh0 = (const float*)d_in[4];
  const float* Wih1 = (const float*)d_in[5];
  const float* Whh1 = (const float*)d_in[6];
  const float* bih1 = (const float*)d_in[7];
  const float* bhh1 = (const float*)d_in[8];
  const float* Wfc  = (const float*)d_in[9];
  const float* bfc  = (const float*)d_in[10];
  float* out = (float*)d_out;

  char* p = (char*)d_ws;
  auto alloc = [&](size_t bytes) -> char* {
    char* r = p; p += (bytes + 255) & ~(size_t)255; return r;
  };
  __bf16* xb    = (__bf16*)alloc((size_t)BTROWS * INPAD * sizeof(__bf16));
  __bf16* wih0b = (__bf16*)alloc((size_t)HD * INPAD * sizeof(__bf16));
  __bf16* whh0b = (__bf16*)alloc((size_t)HD * HD * sizeof(__bf16));
  __bf16* wih1b = (__bf16*)alloc((size_t)HD * HD * sizeof(__bf16));
  __bf16* whh1b = (__bf16*)alloc((size_t)HD * HD * sizeof(__bf16));
  float*  pre   = (float*) alloc((size_t)BTROWS * HD * sizeof(float));
  __bf16* h0b   = (__bf16*)alloc((size_t)BTROWS * HD * sizeof(__bf16));
  __bf16* hlast = (__bf16*)alloc((size_t)BB * HD * sizeof(__bf16));

  // 1) pack inputs/weights to bf16 (pad IN 306 -> 320)
  pack_bf16_pad<<<(BTROWS * INPAD + 255) / 256, 256, 0, stream>>>(x, xb, BTROWS, INP, INPAD);
  pack_bf16_pad<<<(HD * INPAD + 255) / 256, 256, 0, stream>>>(Wih0, wih0b, HD, INP, INPAD);
  pack_bf16_pad<<<(HD * HD + 255) / 256, 256, 0, stream>>>(Whh0, whh0b, HD, HD, HD);
  pack_bf16_pad<<<(HD * HD + 255) / 256, 256, 0, stream>>>(Wih1, wih1b, HD, HD, HD);
  pack_bf16_pad<<<(HD * HD + 255) / 256, 256, 0, stream>>>(Whh1, whh1b, HD, HD, HD);

  dim3 ggrid(BTROWS / 128, HD / 64);   // 500 x 8

  // 2) layer 0: pre = x @ Wih0^T + b ; then scan
  gemm_bias<<<ggrid, 256, 0, stream>>>(xb, wih0b, bih0, bhh0, pre, INPAD);
  rnn_scan<<<BB / 16, 256, 0, stream>>>(pre, whh0b, h0b, /*store_all=*/1);

  // 3) layer 1: pre = h0 @ Wih1^T + b ; then scan (keep only t = T-1)
  gemm_bias<<<ggrid, 256, 0, stream>>>(h0b, wih1b, bih1, bhh1, pre, HD);
  rnn_scan<<<BB / 16, 256, 0, stream>>>(pre, whh1b, hlast, /*store_all=*/0);

  // 4) classifier head + softmax
  fc_softmax<<<1, 128, 0, stream>>>(hlast, Wfc, bfc, out);
}
